// s_gcn_26250840113534
// MI455X (gfx1250) — compile-verified, matrix operand order checked by reading, not run on hardware
//
#include <hip/hip_runtime.h>
#include <cstdint>

// ---------------- problem constants ----------------
#define NCH    64      // channels (in == out)
#define T_ALL  300
#define V_J    25
#define TT     16                // time steps per workgroup tile
#define COLS   (TT * V_J)        // 400 columns per tile
#define COLS_PAD 416
#define CSTR   72                // padded channel stride for [col][c] f16 buffers (144B rows)
#define ASTR   512               // stride for [c][tt*32+v] f16 buffer (v padded 25->32)
#define NT_TILES (COLS / 16)     // 25 WMMA column tiles

typedef __attribute__((ext_vector_type(16))) _Float16 v16h;
typedef __attribute__((ext_vector_type(8)))  _Float16 v8h;
typedef __attribute__((ext_vector_type(8)))  float    v8f;

// 32-byte LDS vector load (all call sites are 16B-aligned by construction)
__device__ __forceinline__ v16h lds_ld16(const _Float16* p) {
    v8h lo = *(const v8h*)p;
    v8h hi = *(const v8h*)(p + 8);
    return __builtin_shufflevector(lo, hi, 0,1,2,3,4,5,6,7,8,9,10,11,12,13,14,15);
}

__device__ __forceinline__ v8f wmma_f16(v16h a, v16h b, v8f c) {
    // D = A(16x32) * B(32x16) + C, fp32 accumulate
    return __builtin_amdgcn_wmma_f32_16x16x32_f16(false, a, false, b, (short)0, c,
                                                  false, false);
}

// One channel-GEMM stage: out[o,col] = act( sum_c W[o,c] * in[c,col] + bias[o] )
// OUT_LAYOUT 0 : store to [o][ASTR] at tt*32+v   (A-operand layout for graph mix)
// OUT_LAYOUT 1 : store to [col][CSTR] at o       (B-operand layout; packed b128 store)
template<int OUT_LAYOUT, bool LEAKY>
__device__ __forceinline__ void channel_stage(const _Float16* __restrict__ inB,
                                              const _Float16* __restrict__ W,
                                              const float*    __restrict__ bias,
                                              _Float16*       __restrict__ outBuf,
                                              int wave, int ln, int lg) {
    // Preload the full 64x64 weight matrix as WMMA A-operands (4 row tiles x 2 k-steps)
    v16h wA[4][2];
#pragma unroll
    for (int mt = 0; mt < 4; ++mt)
#pragma unroll
        for (int ks = 0; ks < 2; ++ks)
            wA[mt][ks] = lds_ld16(W + (mt * 16 + ln) * CSTR + ks * 32 + lg * 16);

    for (int ct = wave; ct < NT_TILES; ct += 8) {
        const int col = ct * 16 + ln;                 // this lane's output column
        // B operand: lane (n = ln) reads K = lg*16..+15 contiguous channels
        v16h b0 = lds_ld16(inB + col * CSTR + lg * 16);        // k = 0..31
        v16h b1 = lds_ld16(inB + col * CSTR + 32 + lg * 16);   // k = 32..63
        const int tt = col / V_J;
        const int v  = col % V_J;
#pragma unroll
        for (int mt = 0; mt < 4; ++mt) {
            v8f acc = {};
            acc = wmma_f16(wA[mt][0], b0, acc);
            acc = wmma_f16(wA[mt][1], b1, acc);
            if (OUT_LAYOUT == 0) {
                // scatter: rows o are ASTR apart (needed as stage-2 A-operand)
#pragma unroll
                for (int r = 0; r < 8; ++r) {
                    const int o = mt * 16 + lg * 8 + r;
                    float t = acc[r] + bias[o];
                    if (LEAKY) t = t > 0.f ? t : 0.01f * t;
                    outBuf[o * ASTR + tt * 32 + v] = (_Float16)t;
                }
            } else {
                // rows o = mt*16+lg*8 .. +7 are consecutive halfwords -> one b128 store
                v8h pk;
#pragma unroll
                for (int r = 0; r < 8; ++r) {
                    const int o = mt * 16 + lg * 8 + r;
                    float t = acc[r] + bias[o];
                    if (LEAKY) t = t > 0.f ? t : 0.01f * t;
                    pk[r] = (_Float16)t;
                }
                *(v8h*)(outBuf + col * CSTR + mt * 16 + lg * 8) = pk;
            }
        }
    }
}

__global__ void __launch_bounds__(256, 2)
sgcn_fused_kernel(const float* __restrict__ x,      const float* __restrict__ A,
                  const float* __restrict__ conv_w, const float* __restrict__ conv_b,
                  const float* __restrict__ w1,     const float* __restrict__ b1,
                  const float* __restrict__ w2,     const float* __restrict__ b2,
                  float* __restrict__ out) {
    extern __shared__ char smem_raw[];
    _Float16* sB   = (_Float16*)smem_raw;            // [COLS_PAD][CSTR]  B-operand buffer
    _Float16* sA   = sB  + COLS_PAD * CSTR;          // [64][ASTR]        A-operand buffer
    _Float16* sW0  = sA  + 64 * ASTR;                // conv_w  [64][CSTR]
    _Float16* sW1  = sW0 + 64 * CSTR;                // w1      [64][CSTR]
    _Float16* sW2  = sW1 + 64 * CSTR;                // w2      [64][CSTR]
    _Float16* sAT  = sW2 + 64 * CSTR;                // A^T padded [32][32]
    float*    sBias = (float*)(sAT + 32 * 32);       // conv_b | b1 | b2

    const int tid   = threadIdx.x;
    const int nIdx  = blockIdx.x;          // sample
    const int t0    = blockIdx.y * TT;     // time-tile origin
    const size_t xBaseN = (size_t)nIdx * NCH * T_ALL * V_J;

    // ---------------- stage 0: stage operands into LDS ----------------
    for (int i = tid; i < NCH * NCH; i += 256) {
        const int o = i >> 6, c = i & 63;
        sW0[o * CSTR + c] = (_Float16)conv_w[i];
        sW1[o * CSTR + c] = (_Float16)w1[i];
        sW2[o * CSTR + c] = (_Float16)w2[i];
    }
    if (tid < NCH) {
        sBias[tid]       = conv_b[tid];
        sBias[64 + tid]  = b1[tid];
        sBias[128 + tid] = b2[tid];
    }
    for (int i = tid; i < 32 * 32; i += 256) {      // A^T, zero-padded 25->32
        const int w = i >> 5, v = i & 31;
        sAT[i] = (_Float16)((v < V_J && w < V_J) ? A[v * V_J + w] : 0.f);
    }
    for (int i = tid; i < NCH * ASTR; i += 256)     // zero v-padding region
        sA[i] = (_Float16)0.f;

    // x tile: (t,v) contiguous and 16B-aligned tile origin -> float4 coalesced loads
    for (int i = tid; i < NCH * (COLS / 4); i += 256) {
        const int c   = i / (COLS / 4);
        const int q   = i % (COLS / 4);
        const int col = q * 4;
        const int g   = t0 * V_J + col;
        const float* src = x + xBaseN + (size_t)c * T_ALL * V_J + g;
        float4 val;
        if (g + 3 < T_ALL * V_J) {
            val = *(const float4*)src;               // global_load_b128
        } else {                                     // tail time-tile: element guard
            val.x = (g + 0 < T_ALL * V_J) ? src[0] : 0.f;
            val.y = (g + 1 < T_ALL * V_J) ? src[1] : 0.f;
            val.z = (g + 2 < T_ALL * V_J) ? src[2] : 0.f;
            val.w = (g + 3 < T_ALL * V_J) ? src[3] : 0.f;
        }
        _Float16* dst = sB + col * CSTR + c;         // transpose scatter into [col][c]
        dst[0 * CSTR] = (_Float16)val.x;
        dst[1 * CSTR] = (_Float16)val.y;
        dst[2 * CSTR] = (_Float16)val.z;
        dst[3 * CSTR] = (_Float16)val.w;
    }
    for (int i = tid; i < NCH * (COLS_PAD - COLS); i += 256) {  // zero pad columns
        const int c = i >> 4, col = COLS + (i & 15);
        sB[col * CSTR + c] = (_Float16)0.f;
    }
    __syncthreads();

    const int wave = tid >> 5;
    const int lane = tid & 31;
    const int ln   = lane & 15;   // N / M index within tile
    const int lg   = lane >> 4;   // K-half / row-half select

    // keep residual x lines warm for stage 4 (x fits in 192MB L2)
    if (tid < NCH)
        __builtin_prefetch(&x[xBaseN + (size_t)tid * T_ALL * V_J + t0 * V_J], 0, 3);

    // ---------------- stage 1: y1 = conv_w @ x + conv_b ----------------
    channel_stage<0, false>(sB, sW0, sBias, sA, wave, ln, lg);
    __syncthreads();

    // ---------------- stage 2: y2[c,t,w] = sum_v y1[c,t,v] * A[v,w] ----------------
    // GEMM rows r=(c,tt) -> tile mt == channel c, M-within-tile == tt; K=v padded to 32
    for (int mt = wave; mt < NCH; mt += 8) {
        v16h a = lds_ld16(sA + mt * ASTR + ln * 32 + lg * 16);   // row tt=ln, v=lg*16..
#pragma unroll
        for (int nt = 0; nt < 2; ++nt) {
            v16h b = lds_ld16(sAT + (nt * 16 + ln) * 32 + lg * 16); // col w, v contiguous
            v8f acc = {};
            acc = wmma_f16(a, b, acc);
            const int w = nt * 16 + ln;
#pragma unroll
            for (int r = 0; r < 8; ++r) {
                const int tt = lg * 8 + r;
                if (w < V_J)
                    sB[(tt * V_J + w) * CSTR + mt] = (_Float16)acc[r];
            }
        }
    }
    __syncthreads();

    // ---------------- stage 3: z1 = leaky(w1 @ y2 + b1) ----------------
    // output in [col][c] layout -> packed ds_store_b128
    channel_stage<1, true>(sB, sW1, sBias + 64, sA, wave, ln, lg);
    __syncthreads();

    // ---------------- stage 4: out = relu( leaky(w2 @ z1 + b2) + x ) ----------------
    {
        v16h wA[4][2];
#pragma unroll
        for (int mt = 0; mt < 4; ++mt)
#pragma unroll
            for (int ks = 0; ks < 2; ++ks)
                wA[mt][ks] = lds_ld16(sW2 + (mt * 16 + ln) * CSTR + ks * 32 + lg * 16);

        for (int ct = wave; ct < NT_TILES; ct += 8) {
            const int col = ct * 16 + ln;
            v16h b0  = lds_ld16(sA + col * CSTR + lg * 16);
            v16h b1v = lds_ld16(sA + col * CSTR + 32 + lg * 16);
            const int tt = col / V_J;
            const int v  = col % V_J;
            const int t  = t0 + tt;
#pragma unroll
            for (int mt = 0; mt < 4; ++mt) {
                v8f acc = {};
                acc = wmma_f16(wA[mt][0], b0, acc);
                acc = wmma_f16(wA[mt][1], b1v, acc);
                if (t < T_ALL) {
#pragma unroll
                    for (int r = 0; r < 8; ++r) {
                        const int o = mt * 16 + lg * 8 + r;
                        const size_t gi = xBaseN + ((size_t)o * T_ALL + t) * V_J + v;
                        float z = acc[r] + sBias[128 + o];
                        z = z > 0.f ? z : 0.01f * z;       // leaky
                        z += x[gi];                        // residual (L2 hit)
                        out[gi] = z > 0.f ? z : 0.f;       // relu (coalesced across lanes)
                    }
                }
            }
        }
    }
}

// tuple output: second element is A passed through
__global__ void copy_A_kernel(const float* __restrict__ A, float* __restrict__ dst) {
    const int i = blockIdx.x * 256 + threadIdx.x;
    if (i < V_J * V_J) dst[i] = A[i];
}

extern "C" void kernel_launch(void* const* d_in, const int* in_sizes, int n_in,
                              void* d_out, int out_size, void* d_ws, size_t ws_size,
                              hipStream_t stream) {
    const float* x      = (const float*)d_in[0];
    const float* A      = (const float*)d_in[1];
    const float* conv_w = (const float*)d_in[2];
    const float* conv_b = (const float*)d_in[3];
    const float* w1     = (const float*)d_in[4];
    const float* b1     = (const float*)d_in[5];
    const float* w2     = (const float*)d_in[6];
    const float* b2     = (const float*)d_in[7];
    float* out = (float*)d_out;

    const size_t shElems = (size_t)COLS_PAD * CSTR + 64 * ASTR + 3 * 64 * CSTR + 32 * 32;
    const size_t shBytes = shElems * 2 + 3 * 64 * sizeof(float);   // ~156 KB (<320KB/WGP)

    dim3 grid(64, (T_ALL + TT - 1) / TT);   // 64 samples x 19 time tiles = 1216 WGs
    sgcn_fused_kernel<<<grid, 256, shBytes, stream>>>(x, A, conv_w, conv_b,
                                                      w1, b1, w2, b2, out);

    // append A to the output tuple tail
    copy_A_kernel<<<3, 256, 0, stream>>>(A, out + (size_t)64 * NCH * T_ALL * V_J);
}
